// RBM_12292196401928
// MI455X (gfx1250) — matrix operand order repaired and checked by reading, not compiled, not run on hardware
//
#include <hip/hip_runtime.h>
#include <stdint.h>

// out = sigmoid(X @ W + bias)
// X: [8192 x 4096] f32 (row-major), W: [4096 x 4096] f32 (row-major),
// bias: [4096] f32, out: [8192 x 4096] f32.
// Compute-bound GEMM -> bf16 WMMA (v_wmma_f32_16x16x32_bf16) with f32 accum.
// A tiles staged via CDNA5 async-to-LDS DMA (ASYNCcnt), converted fp32->bf16
// in the WMMA shadow; B tiles staged with convert+transpose through VGPRs.
// Wave tile 32(M)x64(N) minimizes per-wave A-conversion redundancy and DS issue.

#define M_DIM 8192
#define N_DIM 4096
#define K_DIM 4096

typedef __attribute__((ext_vector_type(4)))  float  f32x4;
typedef __attribute__((ext_vector_type(8)))  float  v8f;
typedef __attribute__((ext_vector_type(4)))  __bf16 v4bf;
typedef __attribute__((ext_vector_type(8)))  __bf16 v8bf;
typedef __attribute__((ext_vector_type(16))) __bf16 v16bf;

static constexpr int BM = 128;    // block tile M
static constexpr int BN = 128;    // block tile N
static constexpr int BK = 32;     // K step == WMMA K
static constexpr int LDAF = 36;   // fp32 A row stride (elems) = 144B, 16B-aligned
static constexpr int LDB  = 40;   // bf16 B row stride (elems) = 80B
static constexpr int KT = K_DIM / BK;  // 128 K-steps

__device__ __forceinline__ void wait_async0() {
#if __has_builtin(__builtin_amdgcn_s_wait_asynccnt)
    __builtin_amdgcn_s_wait_asynccnt(0);
#else
    asm volatile("s_wait_asynccnt 0" ::: "memory");
#endif
}

__global__ __launch_bounds__(256) void rbm_fwd_wmma(
    const float* __restrict__ X, const float* __restrict__ W,
    const float* __restrict__ bias, float* __restrict__ out)
{
    __shared__ __align__(16) float  Asf[2][BM][LDAF];  // fp32 A tile (async DMA dest)
    __shared__ __align__(16) __bf16 Bs [2][BN][LDB];   // bf16 B tile, transposed [N][K]

    const int tid  = threadIdx.x;
    const int lane = tid & 31;
    const int wave = tid >> 5;

    const int m0 = blockIdx.y * BM;
    const int n0 = blockIdx.x * BN;

    // wave tile: 32 (M) x 64 (N) -> 2x4 WMMA tiles; waves arranged 4(M) x 2(N)
    const int wM = (wave & 3) * 32;
    const int wN = (wave >> 2) * 64;

    // A async staging: 4x b128 per thread (contiguous 16B chunks, coalesced)
    const int ac = (tid & 7) * 4;    // float col chunk within BK
    const int ar = tid >> 3;         // 0..31; rows ar + 32*i

    // B staging: 4x4 fp32 micro-tile per thread, transposed bf16 into LDS
    const int kq = (tid >> 5) * 4;   // K base 0..28
    const int nq = (tid & 31) * 4;   // N base 0..124

    const int hsel = lane >> 4;      // 0: lanes 0-15, 1: lanes 16-31
    const int lrow = lane & 15;

    // ---- A: fire-and-forget global -> LDS async copies (ASYNCcnt) ----
    auto asyncA = [&](int buf, int kt) {
        const int kBase = kt * BK;
        #pragma unroll
        for (int i = 0; i < 4; ++i) {
            const float* gp = X + (size_t)(m0 + ar + 32 * i) * K_DIM + kBase + ac;
            unsigned lp = (unsigned)(uintptr_t)&Asf[buf][ar + 32 * i][ac];
            asm volatile("global_load_async_to_lds_b128 %0, %1, off"
                         :: "v"(lp), "v"((uint64_t)(uintptr_t)gp)
                         : "memory");
        }
    };

    // ---- B: coalesced b128 loads + fp32->bf16 convert + transposed store ----
    f32x4 bReg[4];
    auto ldB = [&](int kt) {
        const int kBase = kt * BK;
        #pragma unroll
        for (int i = 0; i < 4; ++i)
            bReg[i] = *(const f32x4*)(W + (size_t)(kBase + kq + i) * N_DIM + n0 + nq);
    };
    auto stB = [&](int buf) {
        #pragma unroll
        for (int j = 0; j < 4; ++j) {
            v4bf h;
            h[0] = (__bf16)bReg[0][j];
            h[1] = (__bf16)bReg[1][j];
            h[2] = (__bf16)bReg[2][j];
            h[3] = (__bf16)bReg[3][j];
            *(v4bf*)&Bs[buf][nq + j][kq] = h;   // 8B ds_store, K-contiguous
        }
    };

    v8f acc[2][4];
    {
        v8f z = {0.f, 0.f, 0.f, 0.f, 0.f, 0.f, 0.f, 0.f};
        #pragma unroll
        for (int mt = 0; mt < 2; ++mt)
            #pragma unroll
            for (int nt = 0; nt < 4; ++nt)
                acc[mt][nt] = z;
    }

    auto compute = [&](int buf) {
        v16bf aF[2], bF[4];
        // A fragment: fp32 from LDS, convert in the WMMA shadow.
        // Lane holds row M=lrow; K {0..7,16..23} (hsel=0) / {8..15,24..31} (hsel=1)
        #pragma unroll
        for (int mt = 0; mt < 2; ++mt) {
            const float* p = &Asf[buf][wM + mt * 16 + lrow][hsel * 8];
            f32x4 a0 = *(const f32x4*)(p);        // K k0..k0+3
            f32x4 a1 = *(const f32x4*)(p + 4);    // K k0+4..k0+7
            f32x4 a2 = *(const f32x4*)(p + 16);   // K k0+16..k0+19
            f32x4 a3 = *(const f32x4*)(p + 20);   // K k0+20..k0+23
            v16bf t;
            #pragma unroll
            for (int i = 0; i < 4; ++i) {
                t[i]      = (__bf16)a0[i];
                t[4 + i]  = (__bf16)a1[i];
                t[8 + i]  = (__bf16)a2[i];
                t[12 + i] = (__bf16)a3[i];
            }
            aF[mt] = t;
        }
        // B fragment: bf16, lane holds col N=lrow; K {0..7,16..23} / {8..15,24..31}
        #pragma unroll
        for (int nt = 0; nt < 4; ++nt) {
            const __bf16* p = &Bs[buf][wN + nt * 16 + lrow][hsel * 8];
            v8bf lo = *(const v8bf*)p;
            v8bf hi = *(const v8bf*)(p + 16);
            bF[nt] = __builtin_shufflevector(lo, hi, 0, 1, 2, 3, 4, 5, 6, 7,
                                             8, 9, 10, 11, 12, 13, 14, 15);
        }
        #pragma unroll
        for (int mt = 0; mt < 2; ++mt)
            #pragma unroll
            for (int nt = 0; nt < 4; ++nt)
                acc[mt][nt] = __builtin_amdgcn_wmma_f32_16x16x32_bf16(
                    false, aF[mt], false, bF[nt], (short)0, acc[mt][nt],
                    false, false);
    };

    // pipeline: stage(0); loop { issue async A(k+1) + load B(k+1) ; wmma(k) ;
    //                            store B(k+1) ; wait asynccnt ; barrier }
    asyncA(0, 0);
    ldB(0);
    stB(0);
    wait_async0();
    __syncthreads();

    #pragma unroll 2
    for (int kt = 0; kt < KT; ++kt) {
        if (kt + 1 < KT) { asyncA((kt + 1) & 1, kt + 1); ldB(kt + 1); }
        compute(kt & 1);
        if (kt + 1 < KT) stB((kt + 1) & 1);
        wait_async0();
        __syncthreads();
    }

    // epilogue: bias + sigmoid; C layout: VGPR r -> row (hsel*8 + r), lane -> col
    #pragma unroll
    for (int nt = 0; nt < 4; ++nt) {
        const int col = n0 + wN + nt * 16 + lrow;
        const float b = bias[col];
        #pragma unroll
        for (int mt = 0; mt < 2; ++mt) {
            const int rowBase = m0 + wM + mt * 16 + hsel * 8;
            #pragma unroll
            for (int r = 0; r < 8; ++r) {
                float v = acc[mt][nt][r] + b;
                out[(size_t)(rowBase + r) * N_DIM + col] =
                    1.0f / (1.0f + __expf(-v));
            }
        }
    }
}

extern "C" void kernel_launch(void* const* d_in, const int* in_sizes, int n_in,
                              void* d_out, int out_size, void* d_ws, size_t ws_size,
                              hipStream_t stream) {
    const float* X    = (const float*)d_in[0];  // [8192 x 4096]
    const float* W    = (const float*)d_in[1];  // [4096 x 4096]
    const float* bias = (const float*)d_in[2];  // [4096]
    float* out = (float*)d_out;                 // [8192 x 4096]

    dim3 grid(N_DIM / BN, M_DIM / BM);          // 32 x 64
    dim3 block(256);
    rbm_fwd_wmma<<<grid, block, 0, stream>>>(X, W, bias, out);
}
